// SimpleSelfAttention_20882130993308
// MI455X (gfx1250) — compile-verified, matrix operand order checked by reading, not run on hardware
//
#include <hip/hip_runtime.h>
#include <hip/hip_bf16.h>

// Problem constants (match reference)
#define D_MODEL 1024
#define BATCH   4
#define SEQ     2048
#define TOK     (BATCH * SEQ)   // 8192 tokens

// GEMM tiling
#define BLK_M 256
#define BLK_N 128
#define BLK_K 32
#define LDS_STRIDE 40           // 32 + 8 pad (bf16 elems) => 80B rows, banks staggered

typedef __attribute__((ext_vector_type(16))) __bf16 v16bf;
typedef __attribute__((ext_vector_type(8)))  float  v8f;

union Frag16 { uint4 q[2]; v16bf v; };   // 32 bytes: one 16x32 bf16 A/B fragment per lane

__device__ __forceinline__ unsigned short f2bf(float f) {
  unsigned u = __float_as_uint(f);
  u += 0x7FFFu + ((u >> 16) & 1u);       // round-to-nearest-even
  return (unsigned short)(u >> 16);
}

// CDNA5 async global->LDS (ASYNCcnt-tracked, no VGPR staging).
// %0 = LDS byte address (low 32 bits of generic shared pointer),
// %1 = 64-bit global address.
__device__ __forceinline__ void async_copy_b128(unsigned lds_byte,
                                                const void* gaddr) {
  asm volatile("global_load_async_to_lds_b128 %0, %1, off"
               :
               : "v"(lds_byte), "v"(gaddr)
               : "memory");
}

__device__ __forceinline__ void wait_async0() {
#if __has_builtin(__builtin_amdgcn_s_wait_asynccnt)
  __builtin_amdgcn_s_wait_asynccnt(0);
#else
  asm volatile("s_wait_asynccnt 0x0" ::: "memory");
#endif
}

// ---------------------------------------------------------------------------
// fp32 -> bf16 bulk convert (vectorized)
// ---------------------------------------------------------------------------
__global__ __launch_bounds__(256) void convert_f32_to_bf16(
    const float4* __restrict__ src, uint2* __restrict__ dst, int n4) {
  int i = blockIdx.x * blockDim.x + threadIdx.x;
  if (i >= n4) return;
  float4 f = src[i];
  uint2 o;
  o.x = (unsigned)f2bf(f.x) | ((unsigned)f2bf(f.y) << 16);
  o.y = (unsigned)f2bf(f.z) | ((unsigned)f2bf(f.w) << 16);
  dst[i] = o;
}

// ---------------------------------------------------------------------------
// NT GEMM:  C[m][n] = scale * sum_k A[m][k]*B[n][k] + bias[n]
//   A: bf16 [M][lda] (K contiguous), B: bf16 [N][ldb] (K contiguous)
//   CMODE 0: C fp32 row-major    (scores, final projection)
//   CMODE 1: C bf16 row-major    (Q, K, attn@V)
//   CMODE 2: C bf16 transposed per batch -> vT[b][n][s]  (V projection)
// Workgroup: 256 thr = 8 waves (4 M x 2 N), wave tile 64x64, 16 WMMA / K-step.
// Double-buffered LDS filled by async global->LDS copies; loads for tile t+1
// overlap the 16 WMMAs of tile t.  Requires K % 64 == 0 (true for all uses).
// ---------------------------------------------------------------------------
template <int CMODE>
__global__ __launch_bounds__(256) void gemm_nt(
    const unsigned short* __restrict__ A, long long strideA, int lda,
    const unsigned short* __restrict__ B, long long strideB, int ldb,
    void* __restrict__ Cv, long long strideC, int ldc,
    const float* __restrict__ bias, float scale, int K) {
  __shared__ unsigned short sA[2][BLK_M * LDS_STRIDE];
  __shared__ unsigned short sB[2][BLK_N * LDS_STRIDE];

  const int tid  = threadIdx.x;
  const int lane = tid & 31;
  const int wave = tid >> 5;
  const int wm   = wave >> 1;        // 0..3 : M direction
  const int wn   = wave & 1;         // 0..1 : N direction
  const int fr   = lane & 15;        // fragment row (m for A, n for B)
  const int fh   = lane >> 4;        // lane half selects K sub-chunks

  const long long batch = blockIdx.z;
  const unsigned short* Ab = A + batch * strideA;
  const unsigned short* Bb = B + batch * strideB;

  const int blockM = blockIdx.x * BLK_M;
  const int blockN = blockIdx.y * BLK_N;

  // global->LDS: 4 threads per 64B row-chunk (32 bf16); each thread owns
  // 4 A-rows (stride 64) and 2 B-rows (stride 64), one 16B chunk per row.
  const int ldRow  = tid >> 2;         // 0..63
  const int ldQuad = (tid & 3) * 8;    // element offset of 16B chunk

  const char* gA = (const char*)(Ab + (long long)(blockM + ldRow) * lda + ldQuad);
  const char* gB = (const char*)(Bb + (long long)(blockN + ldRow) * ldb + ldQuad);
  const long long aRowStep = (long long)64 * lda * 2;   // bytes
  const long long bRowStep = (long long)64 * ldb * 2;

  unsigned ldsA[2][4], ldsB[2][2];
#pragma unroll
  for (int b = 0; b < 2; ++b) {
#pragma unroll
    for (int it = 0; it < 4; ++it)
      ldsA[b][it] =
          (unsigned)(size_t)&sA[b][(ldRow + it * 64) * LDS_STRIDE + ldQuad];
#pragma unroll
    for (int it = 0; it < 2; ++it)
      ldsB[b][it] =
          (unsigned)(size_t)&sB[b][(ldRow + it * 64) * LDS_STRIDE + ldQuad];
  }

  v8f acc[4][4] = {};

  // Issue the 6 async 16B copies for K-step kt into LDS buffer `buf`.
  auto issue = [&](int buf, int kt) {
    const long long ko = (long long)kt * (BLK_K * 2);   // byte offset along K
#pragma unroll
    for (int it = 0; it < 4; ++it)
      async_copy_b128(ldsA[buf][it], gA + it * aRowStep + ko);
#pragma unroll
    for (int it = 0; it < 2; ++it)
      async_copy_b128(ldsB[buf][it], gB + it * bRowStep + ko);
  };

  // 16 WMMAs on LDS buffer `buf` (buf is a compile-time constant at call site).
  auto compute = [&](int buf) {
    Frag16 af[4], bfr[4];
#pragma unroll
    for (int i = 0; i < 4; ++i) {
      int r = (wm * 64 + i * 16 + fr) * LDS_STRIDE;
      af[i].q[0] = *(const uint4*)&sA[buf][r + fh * 8];       // K = fh*8 .. +7
      af[i].q[1] = *(const uint4*)&sA[buf][r + 16 + fh * 8];  // K = 16+fh*8 ..
    }
#pragma unroll
    for (int j = 0; j < 4; ++j) {
      int r = (wn * 64 + j * 16 + fr) * LDS_STRIDE;
      bfr[j].q[0] = *(const uint4*)&sB[buf][r + fh * 8];
      bfr[j].q[1] = *(const uint4*)&sB[buf][r + 16 + fh * 8];
    }
#pragma unroll
    for (int i = 0; i < 4; ++i)
#pragma unroll
      for (int j = 0; j < 4; ++j)
        acc[i][j] = __builtin_amdgcn_wmma_f32_16x16x32_bf16(
            false, af[i].v, false, bfr[j].v, (short)0, acc[i][j], false, false);
  };

  const int nk = K / BLK_K;   // even (32 or 64)
  issue(0, 0);
  wait_async0();
  __syncthreads();

  for (int kt = 0; kt < nk; kt += 2) {
    issue(1, kt + 1);            // kt+1 < nk always (nk even)
    compute(0);
    wait_async0();
    __syncthreads();

    if (kt + 2 < nk) issue(0, kt + 2);
    compute(1);
    wait_async0();
    __syncthreads();
  }

  // Epilogue: C frag layout — lane 0-15: M=vgpr r, lane 16-31: M=8+r; N=lane&15
#pragma unroll
  for (int i = 0; i < 4; ++i) {
    int mBase = blockM + wm * 64 + i * 16 + fh * 8;
#pragma unroll
    for (int j = 0; j < 4; ++j) {
      int n = blockN + wn * 64 + j * 16 + fr;
      float bb = bias ? bias[n] : 0.0f;
#pragma unroll
      for (int r = 0; r < 8; ++r) {
        int m = mBase + r;
        float val = acc[i][j][r] * scale + bb;
        if (CMODE == 0) {
          ((float*)Cv)[batch * strideC + (long long)m * ldc + n] = val;
        } else if (CMODE == 1) {
          ((unsigned short*)Cv)[batch * strideC + (long long)m * ldc + n] =
              f2bf(val);
        } else {
          // vT[b][n][s]: b = m/SEQ, s = m%SEQ
          long long idx = (long long)(m >> 11) * (long long)D_MODEL * SEQ +
                          (long long)n * SEQ + (m & (SEQ - 1));
          ((unsigned short*)Cv)[idx] = f2bf(val);
        }
      }
    }
  }
}

// ---------------------------------------------------------------------------
// Row softmax over SEQ fp32 scores, writing bf16 attn IN PLACE
// (bf16 row occupies first half of the fp32 row; row stride stays 2048 f32)
// ---------------------------------------------------------------------------
__global__ __launch_bounds__(256) void softmax_inplace_bf16(
    float* __restrict__ scores) {
  __shared__ float red[256];
  const long long row = blockIdx.x;
  float* src = scores + row * SEQ;
  unsigned short* dst = (unsigned short*)src;
  const int t = threadIdx.x;

  float v[8];
  float m = -3.402823466e38f;
#pragma unroll
  for (int i = 0; i < 8; ++i) {
    v[i] = src[t + i * 256];
    m = fmaxf(m, v[i]);
  }
  red[t] = m;
  __syncthreads();
  for (int off = 128; off > 0; off >>= 1) {
    if (t < off) red[t] = fmaxf(red[t], red[t + off]);
    __syncthreads();
  }
  m = red[0];
  __syncthreads();

  float s = 0.f;
#pragma unroll
  for (int i = 0; i < 8; ++i) {
    v[i] = __expf(v[i] - m);
    s += v[i];
  }
  red[t] = s;
  __syncthreads();
  for (int off = 128; off > 0; off >>= 1) {
    if (t < off) red[t] += red[t + off];
    __syncthreads();
  }
  float inv = 1.0f / red[0];
#pragma unroll
  for (int i = 0; i < 8; ++i) dst[t + i * 256] = f2bf(v[i] * inv);
}

// ---------------------------------------------------------------------------
// Launcher
// Workspace layout (bytes, 256-aligned). Total required: 142,606,336 B.
//   [0]          xb  : x bf16           16,777,216   (reused for attn@V out)
//   [16777216]   Wq  : bf16              2,097,152
//   [18874368]   Wk  : bf16              2,097,152
//   [20971520]   Wv  : bf16              2,097,152
//   [23068672]   Wp  : bf16              2,097,152
//   [25165824]   q   : bf16             16,777,216
//   [41943040]   k   : bf16             16,777,216
//   [58720256]   vT  : bf16             16,777,216
//   [75497472]   sc  : f32 scores       67,108,864   (softmax rewrites bf16 in place)
// ---------------------------------------------------------------------------
extern "C" void kernel_launch(void* const* d_in, const int* in_sizes, int n_in,
                              void* d_out, int out_size, void* d_ws,
                              size_t ws_size, hipStream_t stream) {
  (void)in_sizes; (void)n_in; (void)out_size; (void)ws_size;
  const float* x  = (const float*)d_in[0];
  const float* Wq = (const float*)d_in[1];
  const float* bq = (const float*)d_in[2];
  const float* Wk = (const float*)d_in[3];
  const float* bk = (const float*)d_in[4];
  const float* Wv = (const float*)d_in[5];
  const float* bv = (const float*)d_in[6];
  const float* Wp = (const float*)d_in[7];
  const float* bp = (const float*)d_in[8];
  float* out = (float*)d_out;

  char* ws = (char*)d_ws;
  unsigned short* xb  = (unsigned short*)(ws + 0);
  unsigned short* wqb = (unsigned short*)(ws + 16777216);
  unsigned short* wkb = (unsigned short*)(ws + 18874368);
  unsigned short* wvb = (unsigned short*)(ws + 20971520);
  unsigned short* wpb = (unsigned short*)(ws + 23068672);
  unsigned short* qb  = (unsigned short*)(ws + 25165824);
  unsigned short* kb  = (unsigned short*)(ws + 41943040);
  unsigned short* vT  = (unsigned short*)(ws + 58720256);
  float*          sc  = (float*)(ws + 75497472);
  unsigned short* oA  = xb;  // reuse: x dead after QKV projections

  // 1) convert x + weights to bf16
  {
    int n4 = TOK * D_MODEL / 4;
    convert_f32_to_bf16<<<(n4 + 255) / 256, 256, 0, stream>>>(
        (const float4*)x, (uint2*)xb, n4);
    int w4 = D_MODEL * D_MODEL / 4;
    dim3 g((w4 + 255) / 256);
    convert_f32_to_bf16<<<g, 256, 0, stream>>>((const float4*)Wq, (uint2*)wqb, w4);
    convert_f32_to_bf16<<<g, 256, 0, stream>>>((const float4*)Wk, (uint2*)wkb, w4);
    convert_f32_to_bf16<<<g, 256, 0, stream>>>((const float4*)Wv, (uint2*)wvb, w4);
    convert_f32_to_bf16<<<g, 256, 0, stream>>>((const float4*)Wp, (uint2*)wpb, w4);
  }

  // 2) Q, K projections (bf16 row-major), V projection (bf16 transposed)
  {
    dim3 grid(TOK / BLK_M, D_MODEL / BLK_N, 1);
    gemm_nt<1><<<grid, 256, 0, stream>>>(xb, 0LL, D_MODEL, wqb, 0LL, D_MODEL,
                                         (void*)qb, 0LL, D_MODEL, bq, 1.0f,
                                         D_MODEL);
    gemm_nt<1><<<grid, 256, 0, stream>>>(xb, 0LL, D_MODEL, wkb, 0LL, D_MODEL,
                                         (void*)kb, 0LL, D_MODEL, bk, 1.0f,
                                         D_MODEL);
    gemm_nt<2><<<grid, 256, 0, stream>>>(xb, 0LL, D_MODEL, wvb, 0LL, D_MODEL,
                                         (void*)vT, 0LL, 0, bv, 1.0f, D_MODEL);
  }

  // 3) scores = (q @ k^T) / sqrt(D), fp32, batched over B
  {
    dim3 grid(SEQ / BLK_M, SEQ / BLK_N, BATCH);
    gemm_nt<0><<<grid, 256, 0, stream>>>(
        qb, (long long)SEQ * D_MODEL, D_MODEL, kb, (long long)SEQ * D_MODEL,
        D_MODEL, (void*)sc, (long long)SEQ * SEQ, SEQ, nullptr, 0.03125f,
        D_MODEL);
  }

  // 4) softmax rows -> bf16 attn in place (row stride 4096 ushorts)
  softmax_inplace_bf16<<<TOK, 256, 0, stream>>>(sc);

  // 5) attn @ V  (A = bf16 attn, lda 4096; B = vT, K-contiguous), bf16 out
  {
    dim3 grid(SEQ / BLK_M, D_MODEL / BLK_N, BATCH);
    gemm_nt<1><<<grid, 256, 0, stream>>>(
        (const unsigned short*)sc, (long long)SEQ * 2 * SEQ, 2 * SEQ, vT,
        (long long)D_MODEL * SEQ, SEQ, (void*)oA, (long long)SEQ * D_MODEL,
        D_MODEL, nullptr, 1.0f, SEQ);
  }

  // 6) final projection -> fp32 d_out
  {
    dim3 grid(TOK / BLK_M, D_MODEL / BLK_N, 1);
    gemm_nt<0><<<grid, 256, 0, stream>>>(oA, 0LL, D_MODEL, wpb, 0LL, D_MODEL,
                                         (void*)out, 0LL, D_MODEL, bp, 1.0f,
                                         D_MODEL);
  }
}
